// Autoencoder_75531294867904
// MI455X (gfx1250) — matrix operand order repaired and checked by reading, not compiled, not run on hardware
//
#include <hip/hip_runtime.h>

typedef __attribute__((ext_vector_type(16))) _Float16 v16h;
typedef __attribute__((ext_vector_type(8)))  _Float16 v8h;
typedef __attribute__((ext_vector_type(8)))  float    v8f;

namespace cfg {
constexpr int kB = 8192, kT = 50, kF = 24;
constexpr int kH1 = 64, kLAT = 32, kH3 = 64;

// LDS layout (units: _Float16 elements). Weight matrices stored transposed
// ([n][k], k padded to a multiple of 32) so WMMA B-fragments are contiguous;
// they are read ONCE into VGPRs and stay resident for the whole kernel.
constexpr int oW1T = 0;               // [256][32]  W1^T (k<24 valid)
constexpr int oU1T = oW1T + 256 * 32; // [256][64]  U1^T
constexpr int oW2T = oU1T + 256 * 64; // [128][64]  W2^T
constexpr int oU2T = oW2T + 128 * 64; // [128][32]  U2^T
constexpr int oW3T = oU2T + 128 * 32; // [256][32]  W3^T
constexpr int oU3T = oW3T + 256 * 32; // [256][64]  U3^T
constexpr int oWdT = oU3T + 256 * 64; // [32][64]   Wd^T (n<24 valid)
constexpr int oH1  = oWdT + 32 * 64;  // [2][16][64]  h1 state (ping-pong)
constexpr int oH2  = oH1 + 2 * 16 * 64; // [2][16][32] h2 state (ping-pong)
constexpr int oH3  = oH2 + 2 * 16 * 32; // [2][16][64] h3 state (ping-pong)
constexpr int kLdsHalves = oH3 + 2 * 16 * 64;
constexpr size_t kLdsBytes = (size_t)kLdsHalves * sizeof(_Float16);
} // namespace cfg

// ---- WMMA fragment loaders -------------------------------------------------
// A-matrix (16xK f16): row m = lane&15; halves 0..7 -> K = kc*32 + 8g + i,
// halves 8..15 -> K = kc*32 + 16 + 8g + i (ISA 16-bit A layout).
__device__ __forceinline__ v16h fragA(const _Float16* X, int m, int kstride,
                                      int kc, int g) {
  const _Float16* p = X + m * kstride + kc * 32;
  v8h lo = *(const v8h*)(p + 8 * g);
  v8h hi = *(const v8h*)(p + 16 + 8 * g);
  v16h r;
#pragma unroll
  for (int i = 0; i < 8; ++i) { r[i] = lo[i]; r[i + 8] = hi[i]; }
  return r;
}

// A-fragment for the x_t tile, built straight from global f32 (row-major
// [B][T][24]); K padded to 32 with zeros. Four (g == 0) or two (g == 1) 16B global loads per lane.
__device__ __forceinline__ v16h fragAx(const float* __restrict__ xr, int g) {
  v16h r;
  float4 a = *(const float4*)(xr + 8 * g);
  float4 b = *(const float4*)(xr + 8 * g + 4);
  r[0] = (_Float16)a.x; r[1] = (_Float16)a.y;
  r[2] = (_Float16)a.z; r[3] = (_Float16)a.w;
  r[4] = (_Float16)b.x; r[5] = (_Float16)b.y;
  r[6] = (_Float16)b.z; r[7] = (_Float16)b.w;
  if (g == 0) {  // K = 16..23 (valid), lanes16-31 cover K = 24..31 (pad)
    float4 c = *(const float4*)(xr + 16);
    float4 d = *(const float4*)(xr + 20);
    r[8]  = (_Float16)c.x; r[9]  = (_Float16)c.y;
    r[10] = (_Float16)c.z; r[11] = (_Float16)c.w;
    r[12] = (_Float16)d.x; r[13] = (_Float16)d.y;
    r[14] = (_Float16)d.z; r[15] = (_Float16)d.w;
  } else {
#pragma unroll
    for (int i = 8; i < 16; ++i) r[i] = (_Float16)0.f;
  }
  return r;
}

// B-matrix (Kx16 f16), stored transposed WT[n][k]: lane column n = lane&15,
// halves 0..7 -> K = kc*32 + 8g + i, halves 8..15 -> K = kc*32 + 16 + 8g + i.
__device__ __forceinline__ v16h fragB(const _Float16* WT, int n, int kstride,
                                      int kc, int g) {
  const _Float16* p = WT + n * kstride + kc * 32;
  v8h lo = *(const v8h*)(p + 8 * g);
  v8h hi = *(const v8h*)(p + 16 + 8 * g);
  v16h r;
#pragma unroll
  for (int i = 0; i < 8; ++i) { r[i] = lo[i]; r[i + 8] = hi[i]; }
  return r;
}

__device__ __forceinline__ v8f wmma_f(v16h a, v16h b, v8f c) {
  return __builtin_amdgcn_wmma_f32_16x16x32_f16(false, a, false, b, (short)0,
                                                c, false, false);
}

__device__ __forceinline__ float sigmoidf_(float x) {
  return 1.0f / (1.0f + __expf(-x));
}
__device__ __forceinline__ v8f splat8(float x) {
  v8f r;
#pragma unroll
  for (int v = 0; v < 8; ++v) r[v] = x;
  return r;
}

// ---- Fused LSTM autoencoder: one 16-row batch tile per workgroup -----------
__global__ __launch_bounds__(64, 1) void lstm_ae_fused(
    const float* __restrict__ x,
    const float* __restrict__ W1, const float* __restrict__ U1, const float* __restrict__ b1,
    const float* __restrict__ W2, const float* __restrict__ U2, const float* __restrict__ b2,
    const float* __restrict__ W3, const float* __restrict__ U3, const float* __restrict__ b3,
    const float* __restrict__ Wd, const float* __restrict__ bd,
    float* __restrict__ out) {
  using namespace cfg;
  extern __shared__ _Float16 lds[];
  _Float16* W1T = lds + oW1T;  _Float16* U1T = lds + oU1T;
  _Float16* W2T = lds + oW2T;  _Float16* U2T = lds + oU2T;
  _Float16* W3T = lds + oW3T;  _Float16* U3T = lds + oU3T;
  _Float16* WdT = lds + oWdT;
  _Float16* h1buf = lds + oH1;   // two 16x64 buffers
  _Float16* h2buf = lds + oH2;   // two 16x32 buffers
  _Float16* h3buf = lds + oH3;   // two 16x64 buffers

  const int tid  = threadIdx.x;      // 0..63 (2 waves)
  const int lane = tid & 31;
  const int wave = tid >> 5;         // each wave owns half the gate columns
  const int g    = (lane >> 4) & 1;  // lane group (ISA fragment layouts)
  const int ln   = lane & 15;
  const int r0   = blockIdx.x * 16;  // batch-tile base row

  // -------- one-time: f16-transpose all weights into LDS --------
  for (int i = tid; i < 256 * 32; i += 64) { int n = i >> 5, k = i & 31;
    W1T[i] = (k < kF) ? (_Float16)W1[k * 256 + n] : (_Float16)0.f; }
  for (int i = tid; i < 256 * 64; i += 64) { int n = i >> 6, k = i & 63;
    U1T[i] = (_Float16)U1[k * 256 + n]; }
  for (int i = tid; i < 128 * 64; i += 64) { int n = i >> 6, k = i & 63;
    W2T[i] = (_Float16)W2[k * 128 + n]; }
  for (int i = tid; i < 128 * 32; i += 64) { int n = i >> 5, k = i & 31;
    U2T[i] = (_Float16)U2[k * 128 + n]; }
  for (int i = tid; i < 256 * 32; i += 64) { int n = i >> 5, k = i & 31;
    W3T[i] = (_Float16)W3[k * 256 + n]; }
  for (int i = tid; i < 256 * 64; i += 64) { int n = i >> 6, k = i & 63;
    U3T[i] = (_Float16)U3[k * 256 + n]; }
  for (int i = tid; i < 32 * 64; i += 64) { int n = i >> 6, k = i & 63;
    WdT[i] = (n < kF) ? (_Float16)Wd[k * kF + n] : (_Float16)0.f; }
  for (int i = tid; i < 16 * 64; i += 64) h1buf[i] = (_Float16)0.f;  // buf 0
  for (int i = tid; i < 16 * 32; i += 64) h2buf[i] = (_Float16)0.f;  // buf 0
  for (int i = tid; i < 16 * 64; i += 64) h3buf[i] = (_Float16)0.f;  // buf 0
  __syncthreads();

  // -------- hoisted biases (uniform per (wave, jc, ln)) --------
  float b1v[2][4], b3v[2][4];
#pragma unroll
  for (int jj = 0; jj < 2; ++jj) {
    int c = (wave + 2 * jj) * 16 + ln;
#pragma unroll
    for (int gt = 0; gt < 4; ++gt) {
      b1v[jj][gt] = b1[gt * kH1 + c];
      b3v[jj][gt] = b3[gt * kH3 + c];
    }
  }
  float b2v[4];
  {
    int c = wave * 16 + ln;
#pragma unroll
    for (int gt = 0; gt < 4; ++gt) b2v[gt] = b2[gt * kLAT + c];
  }
  const int ocol = wave * 16 + ln;                 // dense output column
  const float bdv = (ocol < kF) ? bd[ocol] : 0.f;

  // ==================== ENCODER: LSTM1 -> LSTM2 ====================
  {
    // Loop-invariant weight B-fragments -> VGPRs (never touch LDS again).
    // gate gt occupies column-chunk (jc + gt*H/16).
    v16h bW1[2][4], bU1[2][4][2];
#pragma unroll
    for (int jj = 0; jj < 2; ++jj) {
      int jc = wave + 2 * jj;
#pragma unroll
      for (int gt = 0; gt < 4; ++gt) {
        int n = (jc + 4 * gt) * 16 + ln;
        bW1[jj][gt]    = fragB(W1T, n, 32, 0, g);
        bU1[jj][gt][0] = fragB(U1T, n, 64, 0, g);
        bU1[jj][gt][1] = fragB(U1T, n, 64, 1, g);
      }
    }
    v16h bW2[4][2], bU2[4];
#pragma unroll
    for (int gt = 0; gt < 4; ++gt) {
      int n = (wave + 2 * gt) * 16 + ln;
      bW2[gt][0] = fragB(W2T, n, 64, 0, g);
      bW2[gt][1] = fragB(W2T, n, 64, 1, g);
      bU2[gt]    = fragB(U2T, n, 32, 0, g);
    }

    float c1c[2][8], c2c[8], h1n[2][8];
#pragma unroll
    for (int j = 0; j < 2; ++j)
#pragma unroll
      for (int v = 0; v < 8; ++v) c1c[j][v] = 0.f;
#pragma unroll
    for (int v = 0; v < 8; ++v) c2c[v] = 0.f;

    for (int t = 0; t < kT; ++t) {
      const int p = t & 1;
      _Float16* h1r = h1buf + p * (16 * 64);
      _Float16* h1w = h1buf + (1 - p) * (16 * 64);
      _Float16* h2r = h2buf + p * (16 * 32);
      _Float16* h2w = h2buf + (1 - p) * (16 * 32);

      // A fragments: x_t straight from global, h1_{t-1} from LDS
      v16h ax  = fragAx(x + ((size_t)(r0 + ln) * kT + t) * kF, g);
      v16h ah0 = fragA(h1r, ln, 64, 0, g);
      v16h ah1 = fragA(h1r, ln, 64, 1, g);

#pragma unroll
      for (int jj = 0; jj < 2; ++jj) {
        v8f z[4];
#pragma unroll
        for (int gt = 0; gt < 4; ++gt) {
          z[gt] = splat8(b1v[jj][gt]);
          z[gt] = wmma_f(ax,  bW1[jj][gt],    z[gt]);
          z[gt] = wmma_f(ah0, bU1[jj][gt][0], z[gt]);
          z[gt] = wmma_f(ah1, bU1[jj][gt][1], z[gt]);
        }
#pragma unroll
        for (int v = 0; v < 8; ++v) {
          float iv = sigmoidf_(z[0][v]), fv = sigmoidf_(z[1][v]);
          float gv = fmaxf(z[2][v], 0.f), ov = sigmoidf_(z[3][v]);
          float cn = fv * c1c[jj][v] + iv * gv;
          c1c[jj][v] = cn;
          h1n[jj][v] = ov * fmaxf(cn, 0.f);
        }
      }
      // write h1_t into the OTHER buffer (no conflict with readers of h1r)
#pragma unroll
      for (int jj = 0; jj < 2; ++jj) {
        const int jc = wave + 2 * jj;
#pragma unroll
        for (int v = 0; v < 8; ++v)
          h1w[(8 * g + v) * 64 + jc * 16 + ln] = (_Float16)h1n[jj][v];
      }
      __syncthreads();  // h1_t visible to both waves

      // ---- LSTM2 consumes h1_t immediately (never hits HBM) ----
      {
        v16h a0 = fragA(h1w, ln, 64, 0, g);
        v16h a1 = fragA(h1w, ln, 64, 1, g);
        v16h ah = fragA(h2r, ln, 32, 0, g);
        v8f z[4];
#pragma unroll
        for (int gt = 0; gt < 4; ++gt) {
          z[gt] = splat8(b2v[gt]);
          z[gt] = wmma_f(a0, bW2[gt][0], z[gt]);
          z[gt] = wmma_f(a1, bW2[gt][1], z[gt]);
          z[gt] = wmma_f(ah, bU2[gt],    z[gt]);
        }
        float h2n[8];
#pragma unroll
        for (int v = 0; v < 8; ++v) {
          float iv = sigmoidf_(z[0][v]), fv = sigmoidf_(z[1][v]);
          float gv = fmaxf(z[2][v], 0.f), ov = sigmoidf_(z[3][v]);
          float cn = fv * c2c[v] + iv * gv;
          c2c[v] = cn;
          h2n[v] = ov * fmaxf(cn, 0.f);
        }
#pragma unroll
        for (int v = 0; v < 8; ++v)
          h2w[(8 * g + v) * 32 + wave * 16 + ln] = (_Float16)h2n[v];
        __syncthreads();  // h2_t visible; next step ping-pongs buffers
      }
    }
  }

  // ==================== DECODER: LSTM3 + Dense ====================
  {
    v16h bW3[2][4], bU3[2][4][2];
#pragma unroll
    for (int jj = 0; jj < 2; ++jj) {
      int jc = wave + 2 * jj;
#pragma unroll
      for (int gt = 0; gt < 4; ++gt) {
        int n = (jc + 4 * gt) * 16 + ln;
        bW3[jj][gt]    = fragB(W3T, n, 32, 0, g);
        bU3[jj][gt][0] = fragB(U3T, n, 64, 0, g);
        bU3[jj][gt][1] = fragB(U3T, n, 64, 1, g);
      }
    }
    v16h bWd0 = fragB(WdT, ocol, 64, 0, g);
    v16h bWd1 = fragB(WdT, ocol, 64, 1, g);

    // z = final h2 (RepeatVector input): constant A-fragment for all t
    _Float16* zfin = h2buf + (kT & 1) * (16 * 32);
    v16h az = fragA(zfin, ln, 32, 0, g);

    float c3c[2][8], h3n[2][8];
#pragma unroll
    for (int j = 0; j < 2; ++j)
#pragma unroll
      for (int v = 0; v < 8; ++v) c3c[j][v] = 0.f;

    for (int t = 0; t < kT; ++t) {
      const int p = t & 1;
      _Float16* h3r = h3buf + p * (16 * 64);
      _Float16* h3w = h3buf + (1 - p) * (16 * 64);

      v16h ah0 = fragA(h3r, ln, 64, 0, g);
      v16h ah1 = fragA(h3r, ln, 64, 1, g);
#pragma unroll
      for (int jj = 0; jj < 2; ++jj) {
        v8f z[4];
#pragma unroll
        for (int gt = 0; gt < 4; ++gt) {
          z[gt] = splat8(b3v[jj][gt]);
          z[gt] = wmma_f(az,  bW3[jj][gt],    z[gt]);
          z[gt] = wmma_f(ah0, bU3[jj][gt][0], z[gt]);
          z[gt] = wmma_f(ah1, bU3[jj][gt][1], z[gt]);
        }
#pragma unroll
        for (int v = 0; v < 8; ++v) {
          float iv = sigmoidf_(z[0][v]), fv = sigmoidf_(z[1][v]);
          float gv = fmaxf(z[2][v], 0.f), ov = sigmoidf_(z[3][v]);
          float cn = fv * c3c[jj][v] + iv * gv;
          c3c[jj][v] = cn;
          h3n[jj][v] = ov * fmaxf(cn, 0.f);
        }
      }
#pragma unroll
      for (int jj = 0; jj < 2; ++jj) {
        const int jc = wave + 2 * jj;
#pragma unroll
        for (int v = 0; v < 8; ++v)
          h3w[(8 * g + v) * 64 + jc * 16 + ln] = (_Float16)h3n[jj][v];
      }
      __syncthreads();  // single barrier per decoder step (ping-pong)

      // ---- TimeDistributed Dense: out_t = h3_t @ Wd + bd ----
      {
        v16h a0 = fragA(h3w, ln, 64, 0, g);
        v16h a1 = fragA(h3w, ln, 64, 1, g);
        v8f acc = splat8(bdv);
        acc = wmma_f(a0, bWd0, acc);
        acc = wmma_f(a1, bWd1, acc);
        if (ocol < kF) {
#pragma unroll
          for (int v = 0; v < 8; ++v) {
            int row = 8 * g + v;
            out[((size_t)(r0 + row) * kT + t) * kF + ocol] = acc[v];
          }
        }
      }
      // next iteration's write targets h3r (already fully read before this
      // iteration's barrier), so no extra barrier is needed here
    }
  }
}

extern "C" void kernel_launch(void* const* d_in, const int* in_sizes, int n_in,
                              void* d_out, int out_size, void* d_ws,
                              size_t ws_size, hipStream_t stream) {
  (void)in_sizes; (void)n_in; (void)out_size; (void)d_ws; (void)ws_size;
  const float* x  = (const float*)d_in[0];
  const float* W1 = (const float*)d_in[1];
  const float* U1 = (const float*)d_in[2];
  const float* b1 = (const float*)d_in[3];
  const float* W2 = (const float*)d_in[4];
  const float* U2 = (const float*)d_in[5];
  const float* b2 = (const float*)d_in[6];
  const float* W3 = (const float*)d_in[7];
  const float* U3 = (const float*)d_in[8];
  const float* b3 = (const float*)d_in[9];
  const float* Wd = (const float*)d_in[10];
  const float* bd = (const float*)d_in[11];
  float* out = (float*)d_out;

  // >64KB dynamic LDS opt-in (gfx1250 WGP has 320KB); deterministic,
  // graph-capture safe (no stream work enqueued).
  (void)hipFuncSetAttribute((const void*)lstm_ae_fused,
                            hipFuncAttributeMaxDynamicSharedMemorySize,
                            (int)cfg::kLdsBytes);

  lstm_ae_fused<<<dim3(cfg::kB / 16), dim3(64), cfg::kLdsBytes, stream>>>(
      x, W1, U1, b1, W2, U2, b2, W3, U3, b3, Wd, bd, out);
}
